// TurboQuantLinear_72533407695083
// MI455X (gfx1250) — matrix operand order repaired and checked by reading, not compiled, not run on hardware
//
#include <hip/hip_runtime.h>

// ---- problem constants (from reference) ----
#define OUT_F 4096
#define IN_F  4096
#define NGRP  32
#define GRP   128

// ---- GEMM tiling ----
#define BM 128       // block tile M
#define BN 256       // block tile N
#define BK 32        // K-step (one WMMA K per step)
#define LDT 40       // LDS row stride in bf16 elems (80B: 16B-aligned, bank-conflict-free)

typedef __attribute__((ext_vector_type(16))) __bf16 v16bf;
typedef __attribute__((ext_vector_type(8)))  float  v8f;
typedef __attribute__((ext_vector_type(4)))  int    v4i;

typedef __attribute__((address_space(1))) v4i glb_v4i;
typedef __attribute__((address_space(3))) v4i lds_v4i;

union FragU { uint4 u[2]; v16bf v; };

// ---- CDNA5 async global->LDS path (probe via __has_builtin) ----
#if defined(__has_builtin)
#if __has_builtin(__builtin_amdgcn_global_load_async_to_lds_b128)
#define HAS_ASYNC_LDS 1
#endif
#if __has_builtin(__builtin_amdgcn_s_wait_asynccnt)
#define HAS_WAIT_ASYNC 1
#endif
#endif

__device__ __forceinline__ void async_cp_b128(const unsigned short* g, unsigned short* l) {
#if HAS_ASYNC_LDS
  // global_load_async_to_lds_b128: per-lane 16B global -> LDS, no VGPR data, ASYNCcnt.
  // Builtin signature (from compiler diagnostic): (int4 AS1*, int4 AS3*, Ii offset, Ii cpol)
  __builtin_amdgcn_global_load_async_to_lds_b128(
      (glb_v4i*)(unsigned long long)g,
      (lds_v4i*)(unsigned)(unsigned long long)l,
      0, 0);
#else
  *(uint4*)l = *(const uint4*)g;   // synchronous fallback (same barrier structure)
#endif
}

template <int N>
__device__ __forceinline__ void wait_async() {
#if HAS_WAIT_ASYNC
  __builtin_amdgcn_s_wait_asynccnt((unsigned short)N);
#elif HAS_ASYNC_LDS
  asm volatile("s_wait_asynccnt %0" ::"i"(N) : "memory");
#endif
}

__device__ __forceinline__ unsigned short f2bf(float f) {
  unsigned x = __float_as_uint(f);
  x += 0x7FFFu + ((x >> 16) & 1u);   // round-to-nearest-even
  return (unsigned short)(x >> 16);
}
__device__ __forceinline__ unsigned pk2(float lo, float hi) {
  return (unsigned)f2bf(lo) | ((unsigned)f2bf(hi) << 16);
}

// ---------------------------------------------------------------------------
// Dequant: one wave per (out, group). Lane handles 4 codes; wave-reduce sum(q^2).
// W written row-major [OUT_F][IN_F] bf16 (== column-major B for the GEMM).
// ---------------------------------------------------------------------------
__global__ void __launch_bounds__(256)
tq_dequant_kernel(const float* __restrict__ normsP,
                  const float* __restrict__ normsR,
                  const int*   __restrict__ idxP,
                  const int*   __restrict__ idxR,
                  unsigned short* __restrict__ W) {
  const int gid  = (int)((blockIdx.x * blockDim.x + threadIdx.x) >> 5);  // o*NGRP+g
  const int lane = (int)(threadIdx.x & 31);

  const int4 vp = ((const int4*)idxP)[gid * 32 + lane];
  const int4 vr = ((const int4*)idxR)[gid * 32 + lane];

  const float stp = 2.f / 15.f;   // 4-bit levels: linspace(-1,1,16)
  const float str = 2.f / 3.f;    // 2-bit levels: linspace(-1,1,4)
  float qp[4] = {-1.f + vp.x * stp, -1.f + vp.y * stp,
                 -1.f + vp.z * stp, -1.f + vp.w * stp};
  float qr[4] = {-1.f + vr.x * str, -1.f + vr.y * str,
                 -1.f + vr.z * str, -1.f + vr.w * str};

  float ssp = qp[0]*qp[0] + qp[1]*qp[1] + qp[2]*qp[2] + qp[3]*qp[3];
  float ssr = qr[0]*qr[0] + qr[1]*qr[1] + qr[2]*qr[2] + qr[3]*qr[3];
  #pragma unroll
  for (int off = 16; off > 0; off >>= 1) {
    ssp += __shfl_xor(ssp, off, 32);
    ssr += __shfl_xor(ssr, off, 32);
  }
  const float sp = normsP[gid] * rsqrtf(ssp + 1e-12f);
  const float sr = normsR[gid] * rsqrtf(ssr + 1e-12f);

  uint2 o;
  o.x = pk2(sp * qp[0] + sr * qr[0], sp * qp[1] + sr * qr[1]);
  o.y = pk2(sp * qp[2] + sr * qr[2], sp * qp[3] + sr * qr[3]);
  ((uint2*)W)[gid * 32 + lane] = o;   // element index gid*128 + lane*4
}

// ---------------------------------------------------------------------------
// One-time x f32 -> bf16 conversion (memory-bound: ~192MB traffic ≈ 8us).
// Keeps ALL packing VALU out of the GEMM inner loop.
// ---------------------------------------------------------------------------
__global__ void __launch_bounds__(256)
tq_cvt_x_kernel(const float* __restrict__ X, unsigned short* __restrict__ Xb) {
  const size_t i = ((size_t)blockIdx.x * 256 + threadIdx.x) * 8;
  const float4* p = (const float4*)(X + i);
  const float4 a = p[0], b = p[1];
  uint4 o;
  o.x = pk2(a.x, a.y);
  o.y = pk2(a.z, a.w);
  o.z = pk2(b.x, b.y);
  o.w = pk2(b.z, b.w);
  *(uint4*)(Xb + i) = o;
}

// ---------------------------------------------------------------------------
// GEMM: Y[M, OUT_F] = Xb * W^T + bias, bf16 WMMA, f32 accumulate.
// 256 threads / 8 waves (2 M x 4 N), block tile 128x256, BK=32.
// Each wave: 64x64 = 4x4 WMMA tiles -> 16 wmma per 16 ds_load_b128 per K-step.
// Double-buffered LDS filled by GLOBAL_LOAD_ASYNC_TO_LDS_B128 (zero staging VGPRs).
// ---------------------------------------------------------------------------
__global__ void __launch_bounds__(256, 1)
tq_gemm_kernel(const unsigned short* __restrict__ Xb,
               const unsigned short* __restrict__ W,
               const float* __restrict__ bias,
               float* __restrict__ Y) {
  __shared__ __align__(16) unsigned short sA[2][BM * LDT];   // 2 x 10 KB
  __shared__ __align__(16) unsigned short sB[2][BN * LDT];   // 2 x 20 KB

  const int tid    = (int)threadIdx.x;
  const int wave   = tid >> 5;
  const int lane   = tid & 31;
  const int lanelo = lane & 15;
  const int hi     = lane >> 4;
  const int waveM  = wave >> 2;      // 0..1
  const int waveN  = wave & 3;       // 0..3
  const int n0 = (int)blockIdx.x * BN;
  const int m0 = (int)blockIdx.y * BM;

  // A staging: 2 threads/row, 16 bf16 each. B staging: 1 thread/row, 32 bf16.
  const int arow  = tid >> 1;        // 0..127
  const int ahalf = tid & 1;         // k offset 16*ahalf

  const unsigned short* gA = Xb + (size_t)(m0 + arow) * IN_F + ahalf * 16;
  const unsigned short* gB = W  + (size_t)(n0 + tid)  * IN_F;
  unsigned short* lA = &sA[0][arow * LDT + ahalf * 16];
  unsigned short* lB = &sB[0][tid * LDT];

  // 6 async b128 issues per thread per K-tile => ASYNCcnt += 6 per wave per tile
  auto ISSUE = [&](int kt, int buf) {
    const unsigned short* ga = gA + (size_t)kt * BK;
    const unsigned short* gb = gB + (size_t)kt * BK;
    unsigned short* la = lA + buf * (BM * LDT);
    unsigned short* lb = lB + buf * (BN * LDT);
    async_cp_b128(ga,      la);
    async_cp_b128(ga +  8, la +  8);
    async_cp_b128(gb,      lb);
    async_cp_b128(gb +  8, lb +  8);
    async_cp_b128(gb + 16, lb + 16);
    async_cp_b128(gb + 24, lb + 24);
  };

  const v8f vzero = {0.f, 0.f, 0.f, 0.f, 0.f, 0.f, 0.f, 0.f};
  v8f acc[4][4];
  #pragma unroll
  for (int ms = 0; ms < 4; ++ms)
    #pragma unroll
    for (int ns = 0; ns < 4; ++ns) acc[ms][ns] = vzero;

  const int KT = IN_F / BK;   // 128

  ISSUE(0, 0);
  ISSUE(1, 1);
  wait_async<6>();            // tile 0 landed (in-order completion)
  __syncthreads();

  for (int kt = 0; kt < KT; ++kt) {
    const int cur = kt & 1;

    // A fragments (16x32): lane(lo,hi) -> 16 contiguous K at hi*16
    v16bf aF[4];
    #pragma unroll
    for (int ms = 0; ms < 4; ++ms) {
      const int row = waveM * 64 + ms * 16 + lanelo;
      FragU f;
      f.u[0] = *(const uint4*)&sA[cur][row * LDT + hi * 16];
      f.u[1] = *(const uint4*)&sA[cur][row * LDT + hi * 16 + 8];
      aF[ms] = f.v;
    }
    // B fragments (32x16): lane column N=lanelo, 16 contiguous K at hi*16
    v16bf bF[4];
    #pragma unroll
    for (int ns = 0; ns < 4; ++ns) {
      const int nrow = waveN * 64 + ns * 16 + lanelo;
      const uint4* p = (const uint4*)&sB[cur][nrow * LDT + hi * 16];
      FragU f;
      f.u[0] = p[0];
      f.u[1] = p[1];
      bF[ns] = f.v;
    }

    #pragma unroll
    for (int ms = 0; ms < 4; ++ms)
      #pragma unroll
      for (int ns = 0; ns < 4; ++ns)
        acc[ms][ns] = __builtin_amdgcn_wmma_f32_16x16x32_bf16(
            false, aF[ms], false, bF[ns], (short)0, acc[ms][ns], false, false);

    __syncthreads();                 // all waves done reading buf `cur`
    if (kt + 2 < KT) {
      ISSUE(kt + 2, cur);            // refill the buffer just consumed
      wait_async<6>();               // newest 6 outstanding are kt+2 => kt+1 landed
    } else {
      wait_async<0>();               // drain: tile kt+1 (if any) landed
    }
    __syncthreads();                 // tile kt+1 visible to all waves
  }

  // Epilogue: bias add + store. C/D layout: N=lanelo, M = hi*8 + vgpr index.
  #pragma unroll
  for (int ns = 0; ns < 4; ++ns) {
    const int n = n0 + waveN * 64 + ns * 16 + lanelo;
    const float bn = bias[n];
    #pragma unroll
    for (int ms = 0; ms < 4; ++ms) {
      const int mb = m0 + waveM * 64 + ms * 16 + hi * 8;
      float* yp = Y + (size_t)mb * OUT_F + n;
      #pragma unroll
      for (int v = 0; v < 8; ++v)
        yp[(size_t)v * OUT_F] = acc[ms][ns][v] + bn;
    }
  }
}

// ---------------------------------------------------------------------------
extern "C" void kernel_launch(void* const* d_in, const int* in_sizes, int n_in,
                              void* d_out, int out_size, void* d_ws, size_t ws_size,
                              hipStream_t stream) {
  const float* x      = (const float*)d_in[0];
  const float* normsP = (const float*)d_in[1];
  const float* normsR = (const float*)d_in[2];
  const float* bias   = (const float*)d_in[3];
  const int*   idxP   = (const int*)d_in[4];
  const int*   idxR   = (const int*)d_in[5];
  float* out = (float*)d_out;

  // workspace layout: [0, 32MB) W bf16; [32MB, 96MB) x bf16
  unsigned short* W  = (unsigned short*)d_ws;
  unsigned short* Xb = (unsigned short*)d_ws + (size_t)OUT_F * IN_F;

  const int Mtot    = in_sizes[0] / IN_F;      // 8192
  const int xElems  = in_sizes[0];             // 33,554,432

  // 1) dequantize weights: one wave per (out, group)
  const int groups = OUT_F * NGRP;             // 131072
  tq_dequant_kernel<<<groups / 8, 256, 0, stream>>>(normsP, normsR, idxP, idxR, W);

  // 2) one-time x f32->bf16 conversion (8 elems/thread)
  tq_cvt_x_kernel<<<xElems / (256 * 8), 256, 0, stream>>>(x, Xb);

  // 3) bf16 WMMA GEMM with fused bias
  dim3 grid(OUT_F / BN, Mtot / BM);            // (16, 64)
  tq_gemm_kernel<<<grid, 256, 0, stream>>>(Xb, W, bias, out);
}
